// FormationPredictor_63041529970810
// MI455X (gfx1250) — matrix operand order repaired and checked by reading, not compiled, not run on hardware
//
#include <hip/hip_runtime.h>
#include <hip/hip_bf16.h>

// LSTM fused kernel for MI455X (gfx1250, wave32, WMMA).
// gates_t = [h_{t-1} | x_t] @ [W_hh | W_ih]^T  via v_wmma_f32_16x16x32_bf16,
// entire recurrence resident in LDS; no gx intermediate in HBM.
// Bias folded into WMMA accumulator; activations via v_tanh_f32; all
// elementwise LDS addressing hoisted out of the time loop; 2-wide pairs.

typedef __attribute__((ext_vector_type(16))) __bf16 v16bf;
typedef __attribute__((ext_vector_type(8)))  __bf16 v8bf;
typedef __attribute__((ext_vector_type(2)))  __bf16 v2bf;
typedef __attribute__((ext_vector_type(8)))  float  v8f;

#define B_   4096
#define T_   512
#define I_   2
#define H_   50
#define O_   3
#define G4   200            // 4*H gate columns
#define NTILES 13           // ceil(208/16): gate cols padded to 208
#define BT   16             // batch rows per workgroup (one WMMA M tile)
#define NW   4              // waves per workgroup
#define NTHREADS (NW * 32)
#define ASTR 72             // a_tile row stride in bf16 (64 data + 8 pad; 16B-aligned rows)
#define GSTR 260            // gates row stride in f32 (8B-aligned rows; lane halves on disjoint banks)
#define CSTR 52             // c_sh row stride in f32 (8B-aligned rows)
#define XROW (T_ * I_)      // 1024 elements per batch row of x
#define NPAIR (BT * H_ / 2) // 400 hidden pairs per step
#define KITER 4             // ceil(NPAIR / NTHREADS)

union AFrag { v16bf v; v8bf h[2]; };
union BFrag { v16bf v; __bf16 e[16]; };

__device__ __forceinline__ float wval(const float* __restrict__ Whh,
                                      const float* __restrict__ Wih,
                                      int g, int k) {
  // Column g of the fused weight [W_hh | W_ih | 0-pad], K index k (0..63).
  if (g >= G4) return 0.0f;
  if (k < H_)       return Whh[g * H_ + k];
  if (k < H_ + I_)  return Wih[g * I_ + (k - H_)];
  return 0.0f;
}

// gfx1250 V_TANH_F32 (TRANS op) — confirmed lowering in round 2.
__device__ __forceinline__ float fast_tanh(float x) {
#if __has_builtin(__builtin_amdgcn_tanhf)
  return __builtin_amdgcn_tanhf(x);
#elif __has_builtin(__builtin_amdgcn_tanh_f32)
  return __builtin_amdgcn_tanh_f32(x);
#else
  const float e = __expf(x + x);
  return 1.0f - 2.0f * __builtin_amdgcn_rcpf(e + 1.0f);
#endif
}

__device__ __forceinline__ float fast_sigmoid(float x) {
  // 1 TRANS + 2 VALU (vs exp+rcp = 2 TRANS + 1 VALU): shortens the serial chain.
  return __builtin_fmaf(0.5f, fast_tanh(0.5f * x), 0.5f);
}

__global__ __launch_bounds__(NTHREADS)
void lstm_fused_kernel(const float* __restrict__ x,
                       const float* __restrict__ W_ih,
                       const float* __restrict__ W_hh,
                       const float* __restrict__ b_ih,
                       const float* __restrict__ b_hh,
                       const float* __restrict__ W_fc,
                       const float* __restrict__ b_fc,
                       float* __restrict__ out) {
  __shared__ __align__(16) __bf16 a_tile[BT * ASTR]; //  2.3 KB : [h | x_t | 0] WMMA A tile
  __shared__ __align__(16) __bf16 x_sh[BT * XROW];   // 32.0 KB : whole x timeline, bf16
  __shared__ __align__(16) float  gates[BT * GSTR];  // 16.6 KB : pre-act gates (bias included)
  __shared__ __align__(16) float  c_sh[BT * CSTR];   //  3.3 KB : cell state (f32)

  const int tid   = threadIdx.x;
  const int wave  = __builtin_amdgcn_readfirstlane(tid >> 5);  // SGPR: scalar branches
  const int lane  = tid & 31;
  const int nloc  = lane & 15;
  const bool hihf = lane >= 16;
  const int aoff  = hihf ? 8 : 0;   // A-layout K sub-offset for upper lane half
  const int mbase = hihf ? 8 : 0;   // C-layout M base for upper lane half
  const int b0    = blockIdx.x * BT;

  // ---- One-time: register-resident B fragments of fused W^T (40KB -> L2 hot),
  // plus per-lane gate-column bias (folded into the WMMA accumulator).
  // Dense bf16 B (32x16) layout: lanes 0-15 hold K-chunks {0..7,16..23},
  // lanes 16-31 hold K-chunks {8..15,24..31} (matches A layout).
  v16bf Bf[4][2];
  float bbias[4];
  int cnt = 0;
  #pragma unroll
  for (int i = 0; i < 4; ++i) {
    const int nt = wave + i * NW;        // wave is SGPR -> scalar condition
    if (nt < NTILES) {
      cnt = i + 1;
      const int g = nt * 16 + nloc;
      bbias[i] = (g < G4) ? (b_ih[g] + b_hh[g]) : 0.0f;
      #pragma unroll
      for (int ks = 0; ks < 2; ++ks) {
        BFrag f;
        const int kb = ks * 32 + aoff;
        #pragma unroll
        for (int j = 0; j < 8; ++j) {
          f.e[2 * j]     = (__bf16)wval(W_hh, W_ih, g, kb + 2 * j + (j >= 4 ? 8 : 0));
          f.e[2 * j + 1] = (__bf16)wval(W_hh, W_ih, g, kb + 2 * j + 1 + (j >= 4 ? 8 : 0));
        }
        Bf[i][ks] = f.v;
      }
    } else {
      bbias[i] = 0.0f;
    }
  }

  // ---- Hoisted elementwise addressing: (r,hc) pairs are invariant across t.
  int prow[KITER], phc[KITER];
  bool pact[KITER];
  #pragma unroll
  for (int k = 0; k < KITER; ++k) {
    const int p  = tid + k * NTHREADS;
    pact[k] = (p < NPAIR);
    const int pp = pact[k] ? p : 0;
    prow[k] = pp / (H_ / 2);             // pair row (0..15)
    phc[k]  = (pp % (H_ / 2)) * 2;       // even hidden column (0..48)
  }

  // ---- LDS init: h0 = 0, c0 = 0, preload x (f32 -> bf16, coalesced).
  for (int idx = tid; idx < BT * ASTR; idx += NTHREADS) a_tile[idx] = (__bf16)0.0f;
  for (int idx = tid; idx < BT * CSTR; idx += NTHREADS) c_sh[idx] = 0.0f;
  for (int idx = tid; idx < BT * XROW; idx += NTHREADS) {
    const int r = idx >> 10;             // XROW == 1024
    const int e = idx & (XROW - 1);
    x_sh[idx] = (__bf16)x[(size_t)(b0 + r) * XROW + e];
  }
  __syncthreads();
  if (tid < BT) {                        // inject x_0 (pair) into A tile cols 50,51
    *(v2bf*)&a_tile[tid * ASTR + H_] = *(const v2bf*)&x_sh[tid * XROW];
  }
  __syncthreads();

  // ---- Sequential recurrence: T steps, fully LDS-resident.
  for (int t = 0; t < T_; ++t) {
    // A fragments (16x64 bf16, two K-steps) from LDS.
    // 16-bit A layout: row = lane%16; lanes 0-15 K-chunks {0..7,16..23},
    // lanes 16-31 K-chunks {8..15,24..31}; +32 for k-step 1.
    const int arow = nloc;
    AFrag A0, A1;
    A0.h[0] = *(const v8bf*)&a_tile[arow * ASTR + aoff];
    A0.h[1] = *(const v8bf*)&a_tile[arow * ASTR + 16 + aoff];
    A1.h[0] = *(const v8bf*)&a_tile[arow * ASTR + 32 + aoff];
    A1.h[1] = *(const v8bf*)&a_tile[arow * ASTR + 48 + aoff];

    #pragma unroll
    for (int i = 0; i < 4; ++i) {
      if (i < cnt) {                     // scalar branch; EXEC stays all-1s
        const float bb = bbias[i];
        v8f acc = {bb, bb, bb, bb, bb, bb, bb, bb};
        acc = __builtin_amdgcn_wmma_f32_16x16x32_bf16(
            false, A0.v, false, Bf[i][0], (short)0, acc, false, false);
        acc = __builtin_amdgcn_wmma_f32_16x16x32_bf16(
            false, A1.v, false, Bf[i][1], (short)0, acc, false, false);
        // C layout: N = lane%16, VGPR j -> M = j + (lane<16 ? 0 : 8).
        const int col = (wave + i * NW) * 16 + nloc;
        #pragma unroll
        for (int j = 0; j < 8; ++j)
          gates[(mbase + j) * GSTR + col] = acc[j];
      }
    }
    __syncthreads();

    // Elementwise LSTM update, 2 hidden columns per thread per iteration.
    #pragma unroll
    for (int k = 0; k < KITER; ++k) {
      if (pact[k]) {
        const int r = prow[k], hc = phc[k];
        const float* gr = &gates[r * GSTR + hc];
        const float2 gi = *(const float2*)(gr);            // i gate pair
        const float2 gf = *(const float2*)(gr + H_);       // f gate pair
        const float2 gg = *(const float2*)(gr + 2 * H_);   // g gate pair
        const float2 go = *(const float2*)(gr + 3 * H_);   // o gate pair
        const float2 cc = *(const float2*)&c_sh[r * CSTR + hc];

        const float cn0 = fast_sigmoid(gf.x) * cc.x + fast_sigmoid(gi.x) * fast_tanh(gg.x);
        const float cn1 = fast_sigmoid(gf.y) * cc.y + fast_sigmoid(gi.y) * fast_tanh(gg.y);
        *(float2*)&c_sh[r * CSTR + hc] = make_float2(cn0, cn1);

        v2bf hp;
        hp.x = (__bf16)(fast_sigmoid(go.x) * fast_tanh(cn0));
        hp.y = (__bf16)(fast_sigmoid(go.y) * fast_tanh(cn1));
        *(v2bf*)&a_tile[r * ASTR + hc] = hp;               // h_t back into A tile
      }
    }
    if ((t + 1 < T_) && tid < BT) {      // stage x_{t+1} pair into cols 50,51
      *(v2bf*)&a_tile[tid * ASTR + H_] = *(const v2bf*)&x_sh[tid * XROW + (t + 1) * I_];
    }
    __syncthreads();
  }

  // ---- FC epilogue: out[r, o] = h_last[r, :] . W_fc[o, :] + b_fc[o]  (48 dots of length 50)
  if (tid < BT * O_) {
    const int r = tid / O_;
    const int o = tid - r * O_;
    float s = b_fc[o];
    #pragma unroll 5
    for (int hc = 0; hc < H_; ++hc)
      s += (float)a_tile[r * ASTR + hc] * W_fc[o * H_ + hc];
    out[(size_t)(b0 + r) * O_ + o] = s;
  }
}

extern "C" void kernel_launch(void* const* d_in, const int* in_sizes, int n_in,
                              void* d_out, int out_size, void* d_ws, size_t ws_size,
                              hipStream_t stream) {
  const float* x    = (const float*)d_in[0];
  const float* W_ih = (const float*)d_in[1];
  const float* W_hh = (const float*)d_in[2];
  const float* b_ih = (const float*)d_in[3];
  const float* b_hh = (const float*)d_in[4];
  const float* W_fc = (const float*)d_in[5];
  const float* b_fc = (const float*)d_in[6];
  float* out = (float*)d_out;

  dim3 grid(B_ / BT);        // 256 workgroups
  dim3 block(NTHREADS);      // 128 threads = 4 waves (wave32)
  hipLaunchKernelGGL(lstm_fused_kernel, grid, block, 0, stream,
                     x, W_ih, W_hh, b_ih, b_hh, W_fc, b_fc, out);
}
